// DeformableFeatureAggregation_65077344469610
// MI455X (gfx1250) — matrix operand order repaired and checked by reading, not compiled, not run on hardware
//
#include <hip/hip_runtime.h>
#include <math.h>

#define EMBED 256
#define GROUPS 8
#define LEVELS 4
#define CAMS 6
#define NUM_PTS 13
#define NUM_LEARN 6
#define BS 2
#define NA 900
#define ROWS (BS * NA)            // 1800
#define ROWS_PAD 1808             // 113 * 16
#define NWFC (GROUPS * CAMS * LEVELS * NUM_PTS)   // 2496
#define CLP (CAMS * LEVELS * NUM_PTS)             // 312

typedef __bf16 bf16;
typedef __attribute__((ext_vector_type(16))) __bf16 v16bf;
typedef __attribute__((ext_vector_type(8)))  __bf16 v8bf;
typedef __attribute__((ext_vector_type(8)))  float  v8f;

__constant__ float FIX_SCALE_C[7][3] = {
    {0.f, 0.f, 0.f}, {0.45f, 0.f, 0.f}, {-0.45f, 0.f, 0.f},
    {0.f, 0.45f, 0.f}, {0.f, -0.45f, 0.f}, {0.f, 0.f, 0.45f}, {0.f, 0.f, -0.45f}};

__constant__ int LVL_W[4] = {176, 88, 44, 22};
__constant__ int LVL_H[4] = {64, 32, 16, 8};

// ---------------- K0: conversions ----------------
__global__ void k_feat_bf16(const float* __restrict__ inst,
                            const float* __restrict__ aemb,
                            bf16* __restrict__ out) {
  int i = blockIdx.x * blockDim.x + threadIdx.x;
  if (i >= ROWS_PAD * EMBED) return;
  float v = 0.f;
  if (i < ROWS * EMBED) v = inst[i] + aemb[i];
  out[i] = (bf16)v;
}

__global__ void k_cvt_bf16(const float* __restrict__ src, bf16* __restrict__ out, int n) {
  int i = blockIdx.x * blockDim.x + threadIdx.x;
  if (i >= n) return;
  out[i] = (bf16)src[i];
}

__global__ void k_zero_bf16(bf16* __restrict__ out, int n) {
  int i = blockIdx.x * blockDim.x + threadIdx.x;
  if (i >= n) return;
  out[i] = (bf16)0.f;
}

// ---------------- K1: keypoints + projection (one wave per anchor) ----------------
__global__ void k_keypoints_project(const float* __restrict__ inst,
                                    const float* __restrict__ anchor,
                                    const float* __restrict__ kp_w,
                                    const float* __restrict__ kp_b,
                                    const float* __restrict__ proj,
                                    const float* __restrict__ wh,
                                    float* __restrict__ grid) {
  int wave = (blockIdx.x * blockDim.x + threadIdx.x) >> 5;
  int lane = threadIdx.x & 31;
  if (wave >= ROWS) return;
  int b = wave / NA, n = wave % NA;

  // 18 dot products of length 256, lane-strided then shfl-reduced
  const float* f = inst + (size_t)wave * EMBED;
  float acc[NUM_LEARN * 3];
#pragma unroll
  for (int j = 0; j < NUM_LEARN * 3; j++) acc[j] = 0.f;
  for (int k = lane; k < EMBED; k += 32) {
    float fv = f[k];
#pragma unroll
    for (int j = 0; j < NUM_LEARN * 3; j++) acc[j] += fv * kp_w[j * EMBED + k];
  }
#pragma unroll
  for (int j = 0; j < NUM_LEARN * 3; j++) {
    float s = acc[j];
    for (int off = 16; off > 0; off >>= 1) s += __shfl_xor(s, off, 32);
    acc[j] = s;
  }

  const float* an = anchor + (size_t)wave * 11;
  float cx = an[0], cy = an[1], cz = an[2];
  float sx = expf(an[3]), sy = expf(an[4]), sz = expf(an[5]);
  float siny = an[6], cosy = an[7];

  int p = lane;
  if (p < NUM_PTS) {
    float kx, ky, kz;
    if (p < 7) {
      kx = FIX_SCALE_C[p][0] * sx;
      ky = FIX_SCALE_C[p][1] * sy;
      kz = FIX_SCALE_C[p][2] * sz;
    } else {
      int j = p - 7;
      float lx = 1.f / (1.f + expf(-(acc[j * 3 + 0] + kp_b[j * 3 + 0]))) - 0.5f;
      float ly = 1.f / (1.f + expf(-(acc[j * 3 + 1] + kp_b[j * 3 + 1]))) - 0.5f;
      float lz = 1.f / (1.f + expf(-(acc[j * 3 + 2] + kp_b[j * 3 + 2]))) - 0.5f;
      kx = lx * sx; ky = ly * sy; kz = lz * sz;
    }
    float rx = cosy * kx - siny * ky + cx;
    float ry = siny * kx + cosy * ky + cy;
    float rz = kz + cz;
#pragma unroll
    for (int c = 0; c < CAMS; c++) {
      const float* P = proj + (size_t)(b * CAMS + c) * 16;
      float px = P[0] * rx + P[1] * ry + P[2]  * rz + P[3];
      float py = P[4] * rx + P[5] * ry + P[6]  * rz + P[7];
      float pz = P[8] * rx + P[9] * ry + P[10] * rz + P[11];
      float d = fmaxf(pz, 1e-5f);
      float u = (px / d) / wh[(b * CAMS + c) * 2 + 0];
      float v = (py / d) / wh[(b * CAMS + c) * 2 + 1];
      size_t gi = ((((size_t)(b * CAMS + c)) * NA + n) * NUM_PTS + p) * 2;
      grid[gi + 0] = u * 2.f - 1.f;
      grid[gi + 1] = v * 2.f - 1.f;
    }
  }
}

// ---------------- K2/K5: WMMA bf16 GEMM, C = A * B^T (+bias, +resid) ----------------
// A: [Mtiles*16, K] row-major bf16.  B: [N, K] row-major bf16.
// One wave per 16 x (16*NT) output strip: the A fragment is loaded once per
// K-step and reused by NT accumulating WMMAs (4x less A traffic, 4x WMMA ILP).
// Fragment layouts per CDNA5 ISA 7.12.2:
//   A 16x32: lane l -> M = l&15; half h = l>>4; elems 0..7 -> K = h*8+e, elems 8..15 -> K = 16+h*8+(e-8)
//   B 32x16: lane l -> N = l&15; half h = l>>4; elems 0..7 -> K = h*8+e, elems 8..15 -> K = 16+h*8+(e-8)
//   C/D f32: lane l, reg r -> N = l&15, M = r + 8*(l>>4)
template <int KDIM, int NT>
__global__ void k_wmma_gemm(const bf16* __restrict__ A, const bf16* __restrict__ B,
                            float* __restrict__ C, int Mtiles, int Nstrips,
                            int Ncols, int Mvalid,
                            const float* __restrict__ bias,
                            const float* __restrict__ resid) {
  int strip = (blockIdx.x * blockDim.x + threadIdx.x) >> 5;
  int lane = threadIdx.x & 31;
  if (strip >= Mtiles * Nstrips) return;   // wave-uniform
  int mt = strip / Nstrips, ns = strip % Nstrips;
  int ml = lane & 15, h = lane >> 4;

  const bf16* pa = A + (size_t)(mt * 16 + ml) * KDIM + h * 8;
  const bf16* pb = B + (size_t)(ns * 16 * NT + ml) * KDIM + h * 8;

  v8f acc[NT];
#pragma unroll
  for (int t = 0; t < NT; t++) acc[t] = v8f{};

#pragma unroll
  for (int kk = 0; kk < KDIM; kk += 32) {
    v8bf alo = *(const v8bf*)(pa + kk);        // K = h*8 + [0..8)
    v8bf ahi = *(const v8bf*)(pa + kk + 16);   // K = 16 + h*8 + [0..8)
    v16bf a = __builtin_shufflevector(alo, ahi, 0, 1, 2, 3, 4, 5, 6, 7,
                                      8, 9, 10, 11, 12, 13, 14, 15);
#pragma unroll
    for (int t = 0; t < NT; t++) {
      v8bf blo = *(const v8bf*)(pb + (size_t)t * 16 * KDIM + kk);       // K = h*8 + [0..8)
      v8bf bhi = *(const v8bf*)(pb + (size_t)t * 16 * KDIM + kk + 16);  // K = 16 + h*8 + [0..8)
      v16bf bm = __builtin_shufflevector(blo, bhi, 0, 1, 2, 3, 4, 5, 6, 7,
                                         8, 9, 10, 11, 12, 13, 14, 15);
      acc[t] = __builtin_amdgcn_wmma_f32_16x16x32_bf16(
          false, a, false, bm, (short)0, acc[t], false, false);
    }
  }

#pragma unroll
  for (int t = 0; t < NT; t++) {
    int nn = ns * 16 * NT + t * 16 + ml;
#pragma unroll
    for (int r = 0; r < 8; r++) {
      int m = mt * 16 + r + 8 * h;
      if (m < Mvalid) {
        float v = acc[t][r];
        if (bias) v += bias[nn];
        if (resid) v += resid[(size_t)m * Ncols + nn];
        C[(size_t)m * Ncols + nn] = v;
      }
    }
  }
}

// ---------------- K3: softmax over 312 per (row, group); one wave per (row,g) ----------------
__global__ void k_softmax(float* __restrict__ w) {
  int row = blockIdx.x;
  int g = threadIdx.x >> 5;
  int lane = threadIdx.x & 31;
  float* base = w + (size_t)row * NWFC;

  float v[10];
  int cnt = 0;
  float mx = -1e30f;
  for (int idx = lane; idx < CLP; idx += 32) {
    float x = base[idx * GROUPS + g];
    v[cnt++] = x;
    mx = fmaxf(mx, x);
  }
  for (int off = 16; off > 0; off >>= 1) mx = fmaxf(mx, __shfl_xor(mx, off, 32));
  float sum = 0.f;
  for (int i = 0; i < cnt; i++) { v[i] = expf(v[i] - mx); sum += v[i]; }
  for (int off = 16; off > 0; off >>= 1) sum += __shfl_xor(sum, off, 32);
  float inv = 1.f / sum;
  cnt = 0;
  for (int idx = lane; idx < CLP; idx += 32) base[idx * GROUPS + g] = v[cnt++] * inv;
}

// ---------------- K4: bilinear sample + grouped fusion; one block per anchor ----------------
__global__ void k_sample_fuse(const float* __restrict__ w,
                              const float* __restrict__ grid,
                              const float* __restrict__ fm0, const float* __restrict__ fm1,
                              const float* __restrict__ fm2, const float* __restrict__ fm3,
                              bf16* __restrict__ aggA) {
  __shared__ float w_s[NWFC];
  __shared__ float g_s[CAMS * NUM_PTS * 2];
  int row = blockIdx.x;         // b*NA + n
  int e = threadIdx.x;          // channel 0..255
  int b = row / NA, n = row % NA;

  for (int i = e; i < NWFC; i += EMBED) w_s[i] = w[(size_t)row * NWFC + i];
  for (int i = e; i < CAMS * NUM_PTS * 2; i += EMBED) {
    int c = i / (NUM_PTS * 2);
    int rem = i % (NUM_PTS * 2);
    g_s[i] = grid[((((size_t)(b * CAMS + c)) * NA + n) * NUM_PTS) * 2 + rem];
  }
  __syncthreads();

  const float* fms[4] = {fm0, fm1, fm2, fm3};
  int gidx = e >> 5;            // group = e / (EMBED/GROUPS)
  float acc = 0.f;

  for (int c = 0; c < CAMS; c++) {
    for (int p = 0; p < NUM_PTS; p++) {
      float gx = g_s[(c * NUM_PTS + p) * 2 + 0];
      float gy = g_s[(c * NUM_PTS + p) * 2 + 1];
#pragma unroll
      for (int l = 0; l < LEVELS; l++) {
        int W = LVL_W[l], H = LVL_H[l];
        float x = (gx + 1.f) * (W * 0.5f) - 0.5f;
        float y = (gy + 1.f) * (H * 0.5f) - 0.5f;
        float x0f = floorf(x), y0f = floorf(y);
        int x0 = (int)x0f, y0 = (int)y0f;
        float wx = x - x0f, wy = y - y0f;
        const float* fp = fms[l] + ((size_t)((b * CAMS + c) * EMBED + e)) * H * W;
        float v00 = 0.f, v01 = 0.f, v10 = 0.f, v11 = 0.f;
        bool xi0 = (x0 >= 0) && (x0 < W);
        bool xi1 = (x0 + 1 >= 0) && (x0 + 1 < W);
        bool yi0 = (y0 >= 0) && (y0 < H);
        bool yi1 = (y0 + 1 >= 0) && (y0 + 1 < H);
        if (yi0) {
          const float* rp = fp + (size_t)y0 * W;
          if (xi0) v00 = rp[x0];
          if (xi1) v01 = rp[x0 + 1];
        }
        if (yi1) {
          const float* rp = fp + (size_t)(y0 + 1) * W;
          if (xi0) v10 = rp[x0];
          if (xi1) v11 = rp[x0 + 1];
        }
        float bil = v00 * (1 - wx) * (1 - wy) + v01 * wx * (1 - wy) +
                    v10 * (1 - wx) * wy + v11 * wx * wy;
        float wt = w_s[((c * LEVELS + l) * NUM_PTS + p) * GROUPS + gidx];
        acc += wt * bil;
      }
    }
  }
  aggA[(size_t)row * EMBED + e] = (bf16)acc;
}

// ---------------- launch ----------------
extern "C" void kernel_launch(void* const* d_in, const int* in_sizes, int n_in,
                              void* d_out, int out_size, void* d_ws, size_t ws_size,
                              hipStream_t stream) {
  (void)in_sizes; (void)n_in; (void)out_size; (void)ws_size;
  const float* inst   = (const float*)d_in[0];
  const float* anchor = (const float*)d_in[1];
  const float* aemb   = (const float*)d_in[2];
  const float* fm0    = (const float*)d_in[3];
  const float* fm1    = (const float*)d_in[4];
  const float* fm2    = (const float*)d_in[5];
  const float* fm3    = (const float*)d_in[6];
  const float* proj   = (const float*)d_in[7];
  const float* wh     = (const float*)d_in[8];
  const float* kp_w   = (const float*)d_in[9];
  const float* kp_b   = (const float*)d_in[10];
  const float* wfc_w  = (const float*)d_in[11];
  const float* wfc_b  = (const float*)d_in[12];
  const float* out_w  = (const float*)d_in[13];
  const float* out_b  = (const float*)d_in[14];
  float* out = (float*)d_out;

  char* ws = (char*)d_ws;
  size_t off = 0;
  auto alloc = [&](size_t bytes) -> void* {
    void* p = ws + off;
    off = (off + bytes + 255) & ~(size_t)255;
    return p;
  };
  bf16* featA  = (bf16*)alloc((size_t)ROWS_PAD * EMBED * sizeof(bf16));
  bf16* wfcW   = (bf16*)alloc((size_t)NWFC * EMBED * sizeof(bf16));
  bf16* outW   = (bf16*)alloc((size_t)EMBED * EMBED * sizeof(bf16));
  bf16* aggA   = (bf16*)alloc((size_t)ROWS_PAD * EMBED * sizeof(bf16));
  float* w_raw = (float*)alloc((size_t)ROWS * NWFC * sizeof(float));
  float* grid  = (float*)alloc((size_t)BS * CAMS * NA * NUM_PTS * 2 * sizeof(float));

  // K0: precision conversions / padding
  k_feat_bf16<<<(ROWS_PAD * EMBED + 255) / 256, 256, 0, stream>>>(inst, aemb, featA);
  k_cvt_bf16<<<(NWFC * EMBED + 255) / 256, 256, 0, stream>>>(wfc_w, wfcW, NWFC * EMBED);
  k_cvt_bf16<<<(EMBED * EMBED + 255) / 256, 256, 0, stream>>>(out_w, outW, EMBED * EMBED);
  k_zero_bf16<<<(ROWS_PAD * EMBED + 255) / 256, 256, 0, stream>>>(aggA, ROWS_PAD * EMBED);

  // K1: keypoints + camera projection (one wave per anchor)
  k_keypoints_project<<<(ROWS * 32 + 255) / 256, 256, 0, stream>>>(
      inst, anchor, kp_w, kp_b, proj, wh, grid);

  // K2: wfc logits GEMM [1808x256]x[256x2496] on WMMA bf16, 16x64 strips
  {
    int mt = ROWS_PAD / 16, nstrips = NWFC / 64;   // 113 x 39
    int strips = mt * nstrips;
    k_wmma_gemm<EMBED, 4><<<(strips * 32 + 255) / 256, 256, 0, stream>>>(
        featA, wfcW, w_raw, mt, nstrips, NWFC, ROWS, wfc_b, nullptr);
  }

  // K3: softmax over 312 per group
  k_softmax<<<ROWS, 256, 0, stream>>>(w_raw);

  // K4: bilinear sampling + grouped weighted fusion
  k_sample_fuse<<<ROWS, 256, 0, stream>>>(w_raw, grid, fm0, fm1, fm2, fm3, aggA);

  // K5: output projection GEMM + bias + residual on WMMA bf16, 16x64 strips
  {
    int mt = ROWS_PAD / 16, nstrips = EMBED / 64;  // 113 x 4
    int strips = mt * nstrips;
    k_wmma_gemm<EMBED, 4><<<(strips * 32 + 255) / 256, 256, 0, stream>>>(
        aggA, outW, out, mt, nstrips, EMBED, ROWS, out_b, inst);
  }
}
